// Gdn1d_13408887898867
// MI455X (gfx1250) — matrix operand order repaired and checked by reading, not compiled, not run on hardware
//
#include <hip/hip_runtime.h>
#include <hip/hip_bf16.h>

// GDN: y = x * rsqrt(x^2 @ gamma + beta)
// Core = 65536x1024x1024 GEMM on x^2 -> v_wmma_f32_16x16x32_bf16 (f32 accum),
// double-buffered LDS, gamma pre-converted to bf16 so B staging is a raw
// GLOBAL_LOAD_ASYNC_TO_LDS_B128 DMA overlapped with the WMMAs.

typedef __attribute__((ext_vector_type(16))) __bf16 v16bf;
typedef __attribute__((ext_vector_type(8)))  __bf16 v8bf;
typedef __attribute__((ext_vector_type(4)))  __bf16 v4bf;
typedef __attribute__((ext_vector_type(8)))  float  v8f;

#define BM 128
#define BN 128
#define BK 32
#define ASTR 56                 // A row stride, bf16 elems (112 B rows, 16B aligned)
#define BSTR 136                // B row stride, bf16 elems (272 B rows, 16B aligned)
#define A_TILE_B (BM * ASTR * 2)       // 14336 B
#define B_TILE_B (BK * BSTR * 2)       // 8704 B
#define B_BASE_B (2 * A_TILE_B)        // B buffers after the two A buffers
#define SMEM_B   (2 * A_TILE_B + 2 * B_TILE_B)   // 46080 B

// ---------- gamma f32 -> bf16 pre-convert (runs once per launch) ----------
__global__ __launch_bounds__(256)
void gamma_to_bf16(const float* __restrict__ g, __bf16* __restrict__ gbf) {
    const size_t i = ((size_t)blockIdx.x * 256 + threadIdx.x) * 4;
    float4 f = *(const float4*)(g + i);
    v4bf c;
    c.x = (__bf16)f.x; c.y = (__bf16)f.y; c.z = (__bf16)f.z; c.w = (__bf16)f.w;
    *(v4bf*)(gbf + i) = c;
}

// ---------- staging helpers ----------
__device__ __forceinline__
void stage_a(const float* __restrict__ x, __bf16* __restrict__ As,
             int m0, int k0, int C, int tid) {
    const int arow  = tid >> 1;            // 0..127
    const int ahalf = (tid & 1) * 16;      // 0 or 16 floats
    const float4* src = (const float4*)(x + (size_t)(m0 + arow) * C + k0 + ahalf);
    __bf16* dst = As + arow * ASTR + ahalf;
    #pragma unroll
    for (int i = 0; i < 4; ++i) {
        float4 f = src[i];
        v4bf c;
        c.x = (__bf16)(f.x * f.x);
        c.y = (__bf16)(f.y * f.y);
        c.z = (__bf16)(f.z * f.z);
        c.w = (__bf16)(f.w * f.w);
        *(v4bf*)(dst + i * 4) = c;
    }
}

// Fallback B staging: convert f32 gamma in VALU
__device__ __forceinline__
void stage_b_cvt(const float* __restrict__ gamma, __bf16* __restrict__ Bs,
                 int n0, int k0, int C, int tid) {
    const int brow = tid >> 3;             // 0..31
    const int bseg = (tid & 7) * 16;       // 0..112 floats
    const float4* src = (const float4*)(gamma + (size_t)(k0 + brow) * C + n0 + bseg);
    __bf16* dst = Bs + brow * BSTR + bseg;
    #pragma unroll
    for (int i = 0; i < 4; ++i) {
        float4 f = src[i];
        v4bf c;
        c.x = (__bf16)f.x; c.y = (__bf16)f.y; c.z = (__bf16)f.z; c.w = (__bf16)f.w;
        *(v4bf*)(dst + i * 4) = c;
    }
}

// Async B staging: raw bf16 copy global -> LDS via async DMA (ASYNCcnt)
__device__ __forceinline__
void stage_b_async(const __bf16* __restrict__ gbf, unsigned ldsBase,
                   int n0, int k0, int C, int tid) {
    const int lane16 = tid & 15;           // 16 lanes cover 128 cols (16B each)
    const int rowb   = (tid >> 4) & 15;    // 16 row-groups, 2 rows per thread
    const unsigned long long base = (unsigned long long)gbf;
    #pragma unroll
    for (int i = 0; i < 2; ++i) {
        const int row = rowb + i * 16;
        const unsigned lds  = ldsBase + (unsigned)(row * (BSTR * 2) + lane16 * 16);
        const unsigned goff = (unsigned)(((k0 + row) * C + n0 + lane16 * 8) * 2);
        asm volatile("global_load_async_to_lds_b128 %0, %1, %2"
                     :: "v"(lds), "v"(goff), "s"(base) : "memory");
    }
}

// ---------- fragment build + 8 WMMAs on one K-slab ----------
__device__ __forceinline__
void mma_step(const __bf16* __restrict__ As, const __bf16* __restrict__ Bs,
              int wm, int wn, int lane, v8f acc[2][4]) {
    // A (16x32): lane = M row; lanes 0-15 hold K 0-7 & 16-23, lanes 16-31 hold 8-15 & 24-31
    const int m16   = lane & 15;
    const int khalf = (lane >> 4) * 8;
    v16bf afrag[2];
    #pragma unroll
    for (int mt = 0; mt < 2; ++mt) {
        const __bf16* pa = As + (wm + mt * 16 + m16) * ASTR + khalf;
        v8bf a0 = *(const v8bf*)(pa);
        v8bf a1 = *(const v8bf*)(pa + 16);
        afrag[mt] = __builtin_shufflevector(a0, a1,
            0,1,2,3,4,5,6,7,8,9,10,11,12,13,14,15);
    }
    // B (32x16): lane = N col; lanes 0-15 hold K 0-7 & 16-23, lanes 16-31 hold 8-15 & 24-31
    v16bf bfrag[4];
    #pragma unroll
    for (int nt = 0; nt < 4; ++nt) {
        const __bf16* pb = Bs + khalf * BSTR + wn + nt * 16 + m16;
        #pragma unroll
        for (int s = 0; s < 8; ++s) {
            bfrag[nt][s]     = pb[s * BSTR];
            bfrag[nt][s + 8] = pb[(s + 16) * BSTR];
        }
    }
    #pragma unroll
    for (int mt = 0; mt < 2; ++mt)
        #pragma unroll
        for (int nt = 0; nt < 4; ++nt)
            acc[mt][nt] = __builtin_amdgcn_wmma_f32_16x16x32_bf16(
                false, afrag[mt], false, bfrag[nt],
                (short)0, acc[mt][nt], false, false);
}

// ---------- main kernel ----------
__global__ __launch_bounds__(256)
void gdn_wmma_bf16_kernel(const float* __restrict__ x,
                          const float* __restrict__ gamma,
                          const float* __restrict__ beta,
                          const __bf16* __restrict__ gbf,   // bf16 gamma (or null)
                          float* __restrict__ out,
                          int C, int use_async)
{
    __shared__ __align__(16) unsigned char smem[SMEM_B];   // offset 0 in LDS

    const int tid  = threadIdx.x;
    const int lane = tid & 31;
    const int w    = tid >> 5;
    const int wm   = (w & 3) * 32;
    const int wn   = (w >> 2) * 64;

    const int m0 = blockIdx.x * BM;
    const int n0 = blockIdx.y * BN;

    v8f acc[2][4] = {};

    const int KT = C / BK;
    int cur = 0;

    // prologue: fill buffer 0
    stage_a(x, (__bf16*)(smem + 0 * A_TILE_B), m0, 0, C, tid);
    if (use_async) {
        stage_b_async(gbf, B_BASE_B + 0 * B_TILE_B, n0, 0, C, tid);
        asm volatile("s_wait_asynccnt 0x0" ::: "memory");
    } else {
        stage_b_cvt(gamma, (__bf16*)(smem + B_BASE_B + 0 * B_TILE_B), n0, 0, C, tid);
    }
    __syncthreads();

    for (int kt = 0; kt < KT; ++kt) {
        const int k0   = kt * BK;
        const int nbuf = cur ^ 1;
        const bool more = (kt + 1 < KT);

        if (more) {
            // launch next tile's staging before computing this one
            if (use_async)
                stage_b_async(gbf, B_BASE_B + (unsigned)nbuf * B_TILE_B,
                              n0, k0 + BK, C, tid);
            else
                stage_b_cvt(gamma, (__bf16*)(smem + B_BASE_B + nbuf * B_TILE_B),
                            n0, k0 + BK, C, tid);
            stage_a(x, (__bf16*)(smem + nbuf * A_TILE_B), m0, k0 + BK, C, tid);
            if (kt + 2 < KT)   // speculative prefetch two tiles ahead
                __builtin_prefetch(x + (size_t)(m0 + (tid >> 1)) * C + k0 + 2 * BK, 0, 0);
        }

        mma_step((const __bf16*)(smem + cur * A_TILE_B),
                 (const __bf16*)(smem + B_BASE_B + cur * B_TILE_B),
                 wm, wn, lane, acc);

        if (use_async && more)
            asm volatile("s_wait_asynccnt 0x0" ::: "memory");
        __syncthreads();
        cur = nbuf;
    }

    // epilogue: y = x * rsqrt(acc + beta)
    const int ncol  = lane & 15;
    const int mhalf = (lane >> 4) * 8;
    #pragma unroll
    for (int nt = 0; nt < 4; ++nt) {
        const int col = n0 + wn + nt * 16 + ncol;
        const float bta = beta[col];
        #pragma unroll
        for (int mt = 0; mt < 2; ++mt) {
            const int rbase = m0 + wm + mt * 16 + mhalf;
            v8f a = acc[mt][nt];
            #pragma unroll
            for (int v = 0; v < 8; ++v) {
                const size_t idx = (size_t)(rbase + v) * C + col;
                out[idx] = x[idx] * rsqrtf(a[v] + bta);
            }
        }
    }
}

extern "C" void kernel_launch(void* const* d_in, const int* in_sizes, int n_in,
                              void* d_out, int out_size, void* d_ws, size_t ws_size,
                              hipStream_t stream) {
    const float* x     = (const float*)d_in[0];
    const float* gamma = (const float*)d_in[1];
    const float* beta  = (const float*)d_in[2];
    float* out = (float*)d_out;

    const int C = in_sizes[2];              // 1024
    const int N = in_sizes[0] / C;          // 65536

    const size_t gbf_bytes = (size_t)C * C * sizeof(__bf16);
    const int use_async = (ws_size >= gbf_bytes) ? 1 : 0;
    __bf16* gbf = (__bf16*)d_ws;

    if (use_async) {
        const int total4 = (C * C) / 4;
        gamma_to_bf16<<<total4 / 256, 256, 0, stream>>>(gamma, gbf);
    }

    dim3 grid(N / BM, C / BN);              // (512, 8)
    gdn_wmma_bf16_kernel<<<grid, 256, 0, stream>>>(
        x, gamma, beta, gbf, out, C, use_async);
}
